// ACMnnAttnEmb_11123965297080
// MI455X (gfx1250) — matrix operand order repaired and checked by reading, not compiled, not run on hardware
//
#include <hip/hip_runtime.h>
#include <cstdint>
#include <math.h>

// ---------------- model dims ----------------
constexpr int SEQ   = 4096;
constexpr int DIM   = 768;
constexpr int NH    = 12;
constexpr int DHEAD = 64;
constexpr int FF    = 3072;
constexpr int NL    = 12;
constexpr int BLKQ  = 256;   // query block
constexpr int WIN   = 256;   // one-sided window
constexpr int NBLK  = SEQ / BLKQ;
constexpr int CHK   = 32;    // attention keys per chunk
constexpr float NEGV = -1000000000.0f;

typedef __attribute__((ext_vector_type(16))) __bf16    v16bf;
typedef __attribute__((ext_vector_type(8)))  float     v8f;
typedef __attribute__((ext_vector_type(4)))  uint32_t  v4u;
typedef __attribute__((ext_vector_type(8)))  int       v8i;
typedef __attribute__((ext_vector_type(4)))  int       v4i;

__device__ __forceinline__ uint16_t f2bf(float f) {
  uint32_t u = __float_as_uint(f);
  u += 0x7FFFu + ((u >> 16) & 1u);      // round-to-nearest-even
  return (uint16_t)(u >> 16);
}
__device__ __forceinline__ float bf2f(uint16_t b) {
  return __uint_as_float(((uint32_t)b) << 16);
}

// ---------------- embedding gather ----------------
__global__ void k_embed(const int* __restrict__ ids, const float* __restrict__ we,
                        const float* __restrict__ pe, const float* __restrict__ te,
                        float* __restrict__ x) {
  for (long long i = (long long)blockIdx.x * blockDim.x + threadIdx.x;
       i < (long long)SEQ * DIM; i += (long long)gridDim.x * blockDim.x) {
    int s = (int)(i / DIM), d = (int)(i - (long long)s * DIM);
    x[i] = we[(long long)ids[s] * DIM + d] + pe[(long long)(s + 2) * DIM + d] + te[d];
  }
}

// ---------------- (a [+ b]) -> LayerNorm -> f32 out + bf16 out ----------------
__global__ __launch_bounds__(256) void k_add_ln(const float* __restrict__ a,
                                                const float* __restrict__ b,
                                                const float* __restrict__ g,
                                                const float* __restrict__ bet,
                                                float* __restrict__ outF,
                                                uint16_t* __restrict__ outB) {
  const int row = blockIdx.x, tid = threadIdx.x;
  __shared__ float red[256];
  const float* ar = a + (long long)row * DIM;
  const float* br = b ? b + (long long)row * DIM : nullptr;
  float tl[3]; float s = 0.f;
#pragma unroll
  for (int i = 0; i < 3; i++) {
    int c = tid + i * 256;
    float v = ar[c] + (br ? br[c] : 0.f);
    tl[i] = v; s += v;
  }
  red[tid] = s; __syncthreads();
  for (int st = 128; st > 0; st >>= 1) { if (tid < st) red[tid] += red[tid + st]; __syncthreads(); }
  float mu = red[0] * (1.0f / DIM); __syncthreads();
  float s2 = 0.f;
#pragma unroll
  for (int i = 0; i < 3; i++) { float dv = tl[i] - mu; s2 += dv * dv; }
  red[tid] = s2; __syncthreads();
  for (int st = 128; st > 0; st >>= 1) { if (tid < st) red[tid] += red[tid + st]; __syncthreads(); }
  float inv = rsqrtf(red[0] * (1.0f / DIM) + 1e-5f);
#pragma unroll
  for (int i = 0; i < 3; i++) {
    int c = tid + i * 256;
    float v = (tl[i] - mu) * inv * g[c] + bet[c];
    outF[(long long)row * DIM + c] = v;
    if (outB) outB[(long long)row * DIM + c] = f2bf(v);
  }
}

// ---------------- f32 -> bf16 ----------------
__global__ void k_cvt(const float* __restrict__ in, uint16_t* __restrict__ out, long long n) {
  for (long long i = (long long)blockIdx.x * blockDim.x + threadIdx.x; i < n;
       i += (long long)gridDim.x * blockDim.x)
    out[i] = f2bf(in[i]);
}

// ---------------- exact GELU then f32 -> bf16 ----------------
__global__ void k_gelu_cvt(const float* __restrict__ in, uint16_t* __restrict__ out, long long n) {
  for (long long i = (long long)blockIdx.x * blockDim.x + threadIdx.x; i < n;
       i += (long long)gridDim.x * blockDim.x) {
    float x = in[i];
    float gv = 0.5f * x * (1.0f + erff(x * 0.70710678118654752f));
    out[i] = f2bf(gv);
  }
}

// ---------------- bf16 WMMA GEMM: C[M,N] = A[M,K]*B[K,N] + bias ----------------
// Block tile 128x64, 8 waves (4x2), each wave 32x32 = 2x2 WMMA 16x16x32 tiles.
// A tile staged by the Tensor Data Mover (wave 0 issues a D# with per-row LDS
// padding: 64B rows + 8B pad == 72B pitch). B tile staged manually (transposed).
__global__ __launch_bounds__(256) void k_gemm(const uint16_t* __restrict__ A,
                                              const uint16_t* __restrict__ B,
                                              const float* __restrict__ bias,
                                              float* __restrict__ C,
                                              uint16_t* __restrict__ Cb,
                                              int M, int N, int K) {
  __shared__ alignas(16) uint16_t sA[128 * 36];
  __shared__ alignas(16) uint16_t sB[64 * 36];
  const int tid = threadIdx.x;
  const int lane = tid & 31;
  const int w  = tid >> 5;
  const int wm = w & 3, wn = w >> 2;
  const int hl = lane >> 4;
  const int ln = lane & 15;
  const int m0 = blockIdx.y * 128;
  const int n0 = blockIdx.x * 64;

  v8f zero = {0.f, 0.f, 0.f, 0.f, 0.f, 0.f, 0.f, 0.f};
  v8f acc[2][2];
  acc[0][0] = zero; acc[0][1] = zero; acc[1][0] = zero; acc[1][1] = zero;

  const int brow = tid >> 3;
  const int bcol = (tid & 7) * 8;
  const uint32_t ldsA = (uint32_t)(uintptr_t)&sA[0];  // LDS byte offset (addr[31:0])

  for (int kb = 0; kb < K; kb += 32) {
    // ---- A tile via TDM (wave-uniform branch: only wave 0 executes it) ----
    if (w == 0) {
      uint64_t ga = (uint64_t)(uintptr_t)(A + (size_t)m0 * K + kb);
      v4u g0;
      g0.x = 1u;                                  // count=1, user descriptor
      g0.y = ldsA;                                // lds_addr
      g0.z = (uint32_t)ga;                        // global_addr[31:0]
      g0.w = (uint32_t)(ga >> 32) | (2u << 30);   // global_addr[56:32] | type=2
      v8i g1;
      g1[0] = (int)((1u << 16)    // data_size = 2 bytes
                  | (1u << 20)    // pad_enable
                  | (3u << 22)    // pad_interval: 16 DWORDs (64B) per pad
                  | (1u << 25));  // pad_amount: 2 DWORDs (8B)
      g1[1] = (int)((uint32_t)K << 16);                         // tensor_dim0 lo16
      g1[2] = (int)(((uint32_t)K >> 16) | ((uint32_t)M << 16)); // dim0 hi | dim1 lo
      g1[3] = (int)(((uint32_t)M >> 16) | (32u << 16));         // dim1 hi | tile_dim0=32
      g1[4] = 128;                                              // tile_dim1=128
      g1[5] = K;                                                // tensor_dim0_stride lo32
      g1[6] = 0;
      g1[7] = 0;
      v4i gz = {0, 0, 0, 0};
      v8i gz8 = {0, 0, 0, 0, 0, 0, 0, 0};
      __builtin_amdgcn_tensor_load_to_lds(g0, g1, gz, gz, gz8, 0);
      __builtin_amdgcn_s_wait_tensorcnt(0);
    }
    // ---- B tile staged transposed: sB[n][k] ----
    union { uint4 q; uint16_t s[8]; } ub;
    const uint16_t* bp = B + (long long)(kb + brow) * N + n0 + bcol;
    ub.q = *(const uint4*)bp;
#pragma unroll
    for (int i = 0; i < 8; i++) sB[(bcol + i) * 36 + brow] = ub.s[i];
    if (kb + 32 < K)
      __builtin_prefetch(bp + 32 * N, 0, 1);      // global_prefetch_b8
    __syncthreads();

    union { v16bf v; uint32_t u[8]; } fa[2], fb[2];
#pragma unroll
    for (int mi = 0; mi < 2; mi++) {
      int row = wm * 32 + mi * 16 + ln;
      const uint32_t* pr = (const uint32_t*)&sA[row * 36];
#pragma unroll
      for (int p = 0; p < 8; p++) {
        int kk2 = ((p < 4) ? p : p + 4) + 4 * hl;
        fa[mi].u[p] = pr[kk2];
      }
    }
#pragma unroll
    for (int ni = 0; ni < 2; ni++) {
      int col = wn * 32 + ni * 16 + ln;
      const uint32_t* pc = (const uint32_t*)&sB[col * 36];
#pragma unroll
      for (int p = 0; p < 8; p++) fb[ni].u[p] = pc[((p < 4) ? p : p + 4) + 4 * hl];
    }
#pragma unroll
    for (int mi = 0; mi < 2; mi++)
#pragma unroll
      for (int ni = 0; ni < 2; ni++)
        acc[mi][ni] = __builtin_amdgcn_wmma_f32_16x16x32_bf16(
            false, fa[mi].v, false, fb[ni].v, (short)0, acc[mi][ni], false, false);
    __syncthreads();
  }

#pragma unroll
  for (int mi = 0; mi < 2; mi++) {
#pragma unroll
    for (int ni = 0; ni < 2; ni++) {
      int col = n0 + wn * 32 + ni * 16 + ln;
      float bv = bias[col];
#pragma unroll
      for (int r = 0; r < 8; r++) {
        int rowg = m0 + wm * 32 + mi * 16 + r + 8 * hl;
        float val = acc[mi][ni][r] + bv;
        if (C)  C [(long long)rowg * N + col] = val;
        if (Cb) Cb[(long long)rowg * N + col] = f2bf(val);
      }
    }
  }
}

// ---------------- WMMA flash-attention band kernel ----------------
// grid = NH*NBLK blocks, 256 threads = 8 waves; wave w owns query rows [w*32, w*32+32).
// Thread tid owns row tid for the softmax bookkeeping.
__global__ __launch_bounds__(256) void k_band_attn_wmma(
    const uint16_t* __restrict__ qb, const uint16_t* __restrict__ kbf,
    const uint16_t* __restrict__ vbf,
    const int* __restrict__ amask, const int* __restrict__ gmask,
    float* __restrict__ out) {
  const int h   = blockIdx.x / NBLK;
  const int nb  = blockIdx.x % NBLK;
  const int tid = threadIdx.x;
  const int lane = tid & 31;
  const int w   = tid >> 5;
  const int hl  = lane >> 4;
  const int ln  = lane & 15;
  const int rowbase = w * 32;

  __shared__ alignas(16) uint16_t Kls[CHK * 72];   // [key][d], pitch 72 u16 (36 u32)
  __shared__ alignas(16) uint16_t Vtls[64 * 40];   // [d][key], pitch 40 u16 (20 u32)
  __shared__ float SP[256 * 33];                   // scores f32 / P bf16, pitch 33 f32
  __shared__ float rs0[256];
  __shared__ float rs1[256];
  __shared__ int   validk[CHK];

  union U8 { v16bf v; uint32_t u[8]; };
  v8f zero = {0.f, 0.f, 0.f, 0.f, 0.f, 0.f, 0.f, 0.f};

  // preload Q fragments: 2 m-tiles x 2 K-steps (A 16x32 bf16 layout)
  U8 fq[2][2];
#pragma unroll
  for (int mi = 0; mi < 2; mi++) {
    int qrow = nb * BLKQ + rowbase + mi * 16 + ln;
    const uint32_t* qp = (const uint32_t*)(qb + (size_t)qrow * DIM + h * DHEAD);
#pragma unroll
    for (int ks = 0; ks < 2; ks++)
#pragma unroll
      for (int p = 0; p < 8; p++) {
        int pi = ((p < 4) ? p : p + 4) + 4 * hl;
        fq[mi][ks].u[p] = qp[ks * 16 + pi];
      }
  }

  v8f Oacc[2][4];
#pragma unroll
  for (int mi = 0; mi < 2; mi++)
#pragma unroll
    for (int nj = 0; nj < 4; nj++) Oacc[mi][nj] = zero;
  float m = -3.0e38f, l = 0.f;

  for (int ch = 0; ch < (3 * BLKQ) / CHK; ch++) {
    const int c0 = ch * CHK;
    {   // stage K chunk [32][64] and V chunk transposed [64][32]
      int key = tid >> 3;
      int dp  = (tid & 7) * 8;
      int kidx = nb * BLKQ - BLKQ + c0 + key;
      uint4 kq = make_uint4(0, 0, 0, 0), vq = make_uint4(0, 0, 0, 0);
      if (kidx >= 0 && kidx < SEQ) {
        kq = *(const uint4*)(kbf + (size_t)kidx * DIM + h * DHEAD + dp);
        vq = *(const uint4*)(vbf + (size_t)kidx * DIM + h * DHEAD + dp);
      }
      uint32_t* dk = (uint32_t*)&Kls[key * 72 + dp];
      dk[0] = kq.x; dk[1] = kq.y; dk[2] = kq.z; dk[3] = kq.w;
      union { uint4 q; uint16_t s[8]; } uv; uv.q = vq;
#pragma unroll
      for (int i = 0; i < 8; i++) Vtls[(dp + i) * 40 + key] = uv.s[i];
      if (tid < CHK) {
        int ki = nb * BLKQ - BLKQ + c0 + tid;
        validk[tid] = (ki >= 0 && ki < SEQ) && (amask[ki] > 0) && (gmask[ki] == 0);
      }
    }
    __syncthreads();

    // S = Q K^T  (32 rows x 32 keys per wave)
    v8f sacc[2][2];
    sacc[0][0] = zero; sacc[0][1] = zero; sacc[1][0] = zero; sacc[1][1] = zero;
#pragma unroll
    for (int nj = 0; nj < 2; nj++) {
      int key = nj * 16 + ln;
      const uint32_t* pk = (const uint32_t*)&Kls[key * 72];
      U8 fk0, fk1;
#pragma unroll
      for (int p = 0; p < 8; p++) {
        int pi = ((p < 4) ? p : p + 4) + 4 * hl;
        fk0.u[p] = pk[pi];
        fk1.u[p] = pk[16 + pi];
      }
#pragma unroll
      for (int mi = 0; mi < 2; mi++) {
        sacc[mi][nj] = __builtin_amdgcn_wmma_f32_16x16x32_bf16(
            false, fq[mi][0].v, false, fk0.v, (short)0, sacc[mi][nj], false, false);
        sacc[mi][nj] = __builtin_amdgcn_wmma_f32_16x16x32_bf16(
            false, fq[mi][1].v, false, fk1.v, (short)0, sacc[mi][nj], false, false);
      }
    }
#pragma unroll
    for (int mi = 0; mi < 2; mi++)
#pragma unroll
      for (int nj = 0; nj < 2; nj++)
#pragma unroll
        for (int r = 0; r < 8; r++) {
          int row = rowbase + mi * 16 + r + 8 * hl;
          SP[row * 33 + nj * 16 + ln] = sacc[mi][nj][r] * 0.125f;
        }
    __syncthreads();

    // per-row online softmax for this chunk (thread tid owns row tid)
    float sv[CHK];
    uint32_t okm = 0;
    float mc = -3.0e38f;
#pragma unroll
    for (int j = 0; j < CHK; j++) {
      float s = SP[tid * 33 + j];
      int c = c0 + j;
      int diff = c - BLKQ - tid;
      bool ok = (diff >= -WIN) && (diff <= WIN) && validk[j];
      sv[j] = s;
      if (ok) { okm |= (1u << j); mc = fmaxf(mc, s); }
    }
    float mnew = fmaxf(m, mc);
    float alpha = expf(m - mnew);       // sentinel-sentinel -> exp(0)=1
    __syncthreads();                    // separate f32 reads from bf16 in-place writes
    float psum = 0.f;
    {
      uint16_t* prow = (uint16_t*)&SP[tid * 33];
#pragma unroll
      for (int j = 0; j < CHK; j++) {
        float p = ((okm >> j) & 1u) ? expf(sv[j] - mnew) : 0.f;
        psum += p;
        prow[j] = f2bf(p);
      }
    }
    l = l * alpha + psum;
    m = mnew;
    rs0[tid] = alpha;
    __syncthreads();

    // rescale O, accumulate O += P V
#pragma unroll
    for (int mi = 0; mi < 2; mi++)
#pragma unroll
      for (int r = 0; r < 8; r++) {
        float al = rs0[rowbase + mi * 16 + r + 8 * hl];
#pragma unroll
        for (int nj = 0; nj < 4; nj++) Oacc[mi][nj][r] *= al;
      }
    U8 fp[2];
#pragma unroll
    for (int mi = 0; mi < 2; mi++) {
      int row = rowbase + mi * 16 + ln;
      const uint32_t* pp = (const uint32_t*)&SP[row * 33];
#pragma unroll
      for (int p = 0; p < 8; p++) {
        int pi = ((p < 4) ? p : p + 4) + 4 * hl;
        fp[mi].u[p] = pp[pi];
      }
    }
#pragma unroll
    for (int nj = 0; nj < 4; nj++) {
      int d = nj * 16 + ln;
      const uint32_t* pv = (const uint32_t*)&Vtls[d * 40];
      U8 fv;
#pragma unroll
      for (int p = 0; p < 8; p++) fv.u[p] = pv[((p < 4) ? p : p + 4) + 4 * hl];
#pragma unroll
      for (int mi = 0; mi < 2; mi++)
        Oacc[mi][nj] = __builtin_amdgcn_wmma_f32_16x16x32_bf16(
            false, fp[mi].v, false, fv.v, (short)0, Oacc[mi][nj], false, false);
    }
    __syncthreads();
  }

  // fold in the global-token column (key 0), compute per-row normalizers
  {
    const uint32_t* q32 = (const uint32_t*)(qb + (size_t)(nb * BLKQ + tid) * DIM + h * DHEAD);
    const uint32_t* k32 = (const uint32_t*)(kbf + (size_t)h * DHEAD);   // seq position 0
    float sg = 0.f;
#pragma unroll
    for (int i = 0; i < 32; i++) {
      uint32_t qa = q32[i], ka = k32[i];
      sg += bf2f((uint16_t)qa) * bf2f((uint16_t)ka);
      sg += bf2f((uint16_t)(qa >> 16)) * bf2f((uint16_t)(ka >> 16));
    }
    sg *= 0.125f;
    float mf = fmaxf(m, sg);
    float af = expf(m - mf);
    float pg = expf(sg - mf);
    float lf = l * af + pg;
    rs0[tid] = af / lf;
    rs1[tid] = pg / lf;
  }
  __syncthreads();

#pragma unroll
  for (int mi = 0; mi < 2; mi++)
#pragma unroll
    for (int nj = 0; nj < 4; nj++) {
      int d = nj * 16 + ln;
      float v0d = bf2f(vbf[(size_t)h * DHEAD + d]);    // v, seq position 0
#pragma unroll
      for (int r = 0; r < 8; r++) {
        int row = rowbase + mi * 16 + r + 8 * hl;
        int sq = nb * BLKQ + row;
        out[(size_t)sq * DIM + h * DHEAD + d] =
            Oacc[mi][nj][r] * rs0[row] + rs1[row] * v0d;
      }
    }
}

// ---------------- global-token attention (writes row 0 per head) ----------------
__global__ __launch_bounds__(256) void k_global_attn(const float* __restrict__ qg,
                                                     const float* __restrict__ kg,
                                                     const float* __restrict__ vg,
                                                     const int* __restrict__ amask,
                                                     float* __restrict__ out) {
  const int h = blockIdx.x, tid = threadIdx.x;
  __shared__ float sc[SEQ];
  __shared__ float red[256];
  __shared__ float qrow[64];
  __shared__ float pacc[4][64];
  if (tid < 64) qrow[tid] = qg[h * DHEAD + tid] * 0.125f;
  __syncthreads();
  float lmax = -3.4e38f;
  for (int s0 = tid; s0 < SEQ; s0 += 256) {
    const float* kp = kg + (long long)s0 * DIM + h * DHEAD;
    float s = 0.f;
#pragma unroll
    for (int d = 0; d < 64; d++) s += qrow[d] * kp[d];
    if (!(amask[s0] > 0)) s = NEGV;
    sc[s0] = s;
    lmax = fmaxf(lmax, s);
  }
  red[tid] = lmax; __syncthreads();
  for (int st = 128; st > 0; st >>= 1) { if (tid < st) red[tid] = fmaxf(red[tid], red[tid + st]); __syncthreads(); }
  float mx = red[0]; __syncthreads();
  float lsum = 0.f;
  for (int s0 = tid; s0 < SEQ; s0 += 256) {
    float p = expf(sc[s0] - mx);
    sc[s0] = p; lsum += p;
  }
  red[tid] = lsum; __syncthreads();
  for (int st = 128; st > 0; st >>= 1) { if (tid < st) red[tid] += red[tid + st]; __syncthreads(); }
  float inv = 1.f / red[0]; __syncthreads();
  int d = tid & 63, part = tid >> 6;
  float a = 0.f;
  for (int s0 = part * 1024; s0 < (part + 1) * 1024; s0++)
    a += sc[s0] * vg[(long long)s0 * DIM + h * DHEAD + d];
  pacc[part][d] = a;
  __syncthreads();
  if (tid < 64)
    out[h * DHEAD + tid] = (pacc[0][tid] + pacc[1][tid] + pacc[2][tid] + pacc[3][tid]) * inv;
}

// ---------------- tiny classification head (single block) ----------------
__global__ __launch_bounds__(256) void k_head(const float* __restrict__ x,
    const float* __restrict__ Wd, const float* __restrict__ bd,
    const float* __restrict__ Wcls, const float* __restrict__ bcls,
    const float* __restrict__ mixw,
    const float* __restrict__ mg, const float* __restrict__ mb,
    const float* __restrict__ Wh, const float* __restrict__ bh,
    const float* __restrict__ Wop, const float* __restrict__ bop,
    float* __restrict__ outy) {
  __shared__ float cls[DIM];
  __shared__ float hbuf[128];
  __shared__ float y2[2];
  __shared__ float lg[2];
  const int tid = threadIdx.x;
  for (int o = tid; o < DIM; o += 256) {
    float a = bd[o];
    for (int kk = 0; kk < DIM; kk++) a += x[kk] * Wd[kk * DIM + o];
    cls[o] = tanhf(a);
  }
  __syncthreads();
  if (tid < 2) {
    float a = bcls[tid];
    for (int kk = 0; kk < DIM; kk++) a += cls[kk] * Wcls[kk * 2 + tid];
    lg[tid] = a;
  }
  __syncthreads();
  if (tid == 0) {
    float mx = mixw[0];
    for (int i = 1; i < NL; i++) mx = fmaxf(mx, mixw[i]);
    float ssum = 0.f;
    for (int i = 0; i < NL; i++) ssum += expf(mixw[i] - mx);
    float tot = 0.f;
    for (int i = 0; i < NL; i++) tot += expf(mixw[i] - mx) / ssum;
    float m0 = lg[0] * tot, m1 = lg[1] * tot;
    float mu = 0.5f * (m0 + m1);
    float va = 0.5f * ((m0 - mu) * (m0 - mu) + (m1 - mu) * (m1 - mu));
    float inv = rsqrtf(va + 1e-5f);
    y2[0] = (m0 - mu) * inv * mg[0] + mb[0];
    y2[1] = (m1 - mu) * inv * mg[1] + mb[1];
  }
  __syncthreads();
  if (tid < 128)
    hbuf[tid] = fmaxf(bh[tid] + y2[0] * Wh[tid] + y2[1] * Wh[128 + tid], 0.f);
  __syncthreads();
  if (tid < 2) {
    float a = bop[tid];
    for (int j = 0; j < 128; j++) a += hbuf[j] * Wop[j * 2 + tid];
    outy[tid] = a;
  }
}

// ---------------- host orchestration ----------------
extern "C" void kernel_launch(void* const* d_in, const int* in_sizes, int n_in,
                              void* d_out, int out_size, void* d_ws, size_t ws_size,
                              hipStream_t stream) {
  const int*   ids   = (const int*)d_in[0];
  const int*   amask = (const int*)d_in[1];
  const int*   gmask = (const int*)d_in[2];
  const float* we    = (const float*)d_in[3];
  const float* pe    = (const float*)d_in[4];
  const float* te    = (const float*)d_in[5];
  const float* eg    = (const float*)d_in[6];
  const float* ebta  = (const float*)d_in[7];
  const float* Wq    = (const float*)d_in[8];
  const float* Wk    = (const float*)d_in[9];
  const float* Wv    = (const float*)d_in[10];
  const float* Wqg   = (const float*)d_in[11];
  const float* Wkg   = (const float*)d_in[12];
  const float* Wvg   = (const float*)d_in[13];
  const float* Wo    = (const float*)d_in[14];
  const float* W1    = (const float*)d_in[15];
  const float* W2    = (const float*)d_in[16];
  const float* bq    = (const float*)d_in[17];
  const float* bk    = (const float*)d_in[18];
  const float* bv    = (const float*)d_in[19];
  const float* bqg   = (const float*)d_in[20];
  const float* bkg   = (const float*)d_in[21];
  const float* bvg   = (const float*)d_in[22];
  const float* bo    = (const float*)d_in[23];
  const float* b2    = (const float*)d_in[24];
  const float* b1    = (const float*)d_in[25];
  const float* ln1g  = (const float*)d_in[26];
  const float* ln1b  = (const float*)d_in[27];
  const float* ln2g  = (const float*)d_in[28];
  const float* ln2b  = (const float*)d_in[29];
  const float* Wd    = (const float*)d_in[30];
  const float* bd    = (const float*)d_in[31];
  const float* Wcls  = (const float*)d_in[32];
  const float* bcls  = (const float*)d_in[33];
  const float* mixw  = (const float*)d_in[34];
  const float* mlng  = (const float*)d_in[35];
  const float* mlnb  = (const float*)d_in[36];
  const float* Wh    = (const float*)d_in[37];
  const float* bh    = (const float*)d_in[38];
  const float* Wop   = (const float*)d_in[39];
  const float* bop   = (const float*)d_in[40];
  (void)in_sizes; (void)n_in; (void)out_size; (void)ws_size;

  char* wsb = (char*)d_ws;
  size_t off = 0;
  auto alloc = [&](size_t bytes) -> void* {
    void* p = wsb + off;
    off += (bytes + 255) & ~(size_t)255;
    return p;
  };
  const size_t SD = (size_t)SEQ * DIM, SF = (size_t)SEQ * FF;
  float*    x     = (float*)   alloc(SD * 4);
  uint16_t* xb    = (uint16_t*)alloc(SD * 2);
  uint16_t* qbf   = (uint16_t*)alloc(SD * 2);
  uint16_t* kbf   = (uint16_t*)alloc(SD * 2);
  uint16_t* vbf   = (uint16_t*)alloc(SD * 2);
  float*    qgf   = (float*)   alloc(SD * 4);
  float*    kgf   = (float*)   alloc(SD * 4);
  float*    vgf   = (float*)   alloc(SD * 4);
  float*    attn  = (float*)   alloc(SD * 4);
  uint16_t* attnb = (uint16_t*)alloc(SD * 2);
  float*    delta = (float*)   alloc(SD * 4);
  float*    hb32  = (float*)   alloc(SF * 4);
  uint16_t* hb16  = (uint16_t*)alloc(SF * 2);
  uint16_t* wb    = (uint16_t*)alloc((size_t)DIM * FF * 2);

  const long long DD = (long long)DIM * DIM;
  const long long DF = (long long)DIM * FF;
  dim3 t256(256);
  dim3 gDD(DIM / 64, SEQ / 128);
  dim3 gDF(FF  / 64, SEQ / 128);

  k_embed<<<dim3(2048), t256, 0, stream>>>(ids, we, pe, te, x);
  k_add_ln<<<dim3(SEQ), t256, 0, stream>>>(x, nullptr, eg, ebta, x, xb);

  for (int l = 0; l < NL; l++) {
    const float* Wq_l  = Wq  + (size_t)l * DD;
    const float* Wk_l  = Wk  + (size_t)l * DD;
    const float* Wv_l  = Wv  + (size_t)l * DD;
    const float* Wqg_l = Wqg + (size_t)l * DD;
    const float* Wkg_l = Wkg + (size_t)l * DD;
    const float* Wvg_l = Wvg + (size_t)l * DD;
    const float* Wo_l  = Wo  + (size_t)l * DD;
    const float* W1_l  = W1  + (size_t)l * DF;
    const float* W2_l  = W2  + (size_t)l * DF;

    // band-attention projections -> bf16 directly
    k_cvt<<<dim3(1024), t256, 0, stream>>>(Wq_l, wb, DD);
    k_gemm<<<gDD, t256, 0, stream>>>(xb, wb, bq + l * DIM, nullptr, qbf, SEQ, DIM, DIM);
    k_cvt<<<dim3(1024), t256, 0, stream>>>(Wk_l, wb, DD);
    k_gemm<<<gDD, t256, 0, stream>>>(xb, wb, bk + l * DIM, nullptr, kbf, SEQ, DIM, DIM);
    k_cvt<<<dim3(1024), t256, 0, stream>>>(Wv_l, wb, DD);
    k_gemm<<<gDD, t256, 0, stream>>>(xb, wb, bv + l * DIM, nullptr, vbf, SEQ, DIM, DIM);
    // global-attention projections (f32 for the scalar kernel)
    k_cvt<<<dim3(1024), t256, 0, stream>>>(Wqg_l, wb, DD);
    k_gemm<<<gDD, t256, 0, stream>>>(xb, wb, bqg + l * DIM, qgf, nullptr, SEQ, DIM, DIM);
    k_cvt<<<dim3(1024), t256, 0, stream>>>(Wkg_l, wb, DD);
    k_gemm<<<gDD, t256, 0, stream>>>(xb, wb, bkg + l * DIM, kgf, nullptr, SEQ, DIM, DIM);
    k_cvt<<<dim3(1024), t256, 0, stream>>>(Wvg_l, wb, DD);
    k_gemm<<<gDD, t256, 0, stream>>>(xb, wb, bvg + l * DIM, vgf, nullptr, SEQ, DIM, DIM);

    k_band_attn_wmma<<<dim3(NH * NBLK), t256, 0, stream>>>(qbf, kbf, vbf, amask, gmask, attn);
    k_global_attn<<<dim3(NH), t256, 0, stream>>>(qgf, kgf, vgf, amask, attn);

    k_cvt<<<dim3(2048), t256, 0, stream>>>(attn, attnb, (long long)SD);
    k_cvt<<<dim3(1024), t256, 0, stream>>>(Wo_l, wb, DD);
    k_gemm<<<gDD, t256, 0, stream>>>(attnb, wb, bo + l * DIM, delta, nullptr, SEQ, DIM, DIM);
    k_add_ln<<<dim3(SEQ), t256, 0, stream>>>(x, delta, ln1g + l * DIM, ln1b + l * DIM, x, xb);

    k_cvt<<<dim3(2048), t256, 0, stream>>>(W1_l, wb, DF);
    k_gemm<<<gDF, t256, 0, stream>>>(xb, wb, b1 + l * FF, hb32, nullptr, SEQ, FF, DIM);
    k_gelu_cvt<<<dim3(4096), t256, 0, stream>>>(hb32, hb16, (long long)SF);
    k_cvt<<<dim3(2048), t256, 0, stream>>>(W2_l, wb, DF);
    k_gemm<<<gDD, t256, 0, stream>>>(hb16, wb, b2 + l * DIM, delta, nullptr, SEQ, DIM, FF);
    k_add_ln<<<dim3(SEQ), t256, 0, stream>>>(x, delta, ln2g + l * DIM, ln2b + l * DIM, x, xb);
  }

  k_head<<<dim3(1), t256, 0, stream>>>(x, Wd, bd, Wcls, bcls, mixw,
                                       mlng, mlnb, Wh, bh, Wop, bop, (float*)d_out);
}
